// AtlasFreeBrainTransformer_2293512536665
// MI455X (gfx1250) — matrix operand order repaired and checked by reading, not compiled, not run on hardware
//
#include <hip/hip_runtime.h>
#include <hip/hip_bf16.h>
#include <math.h>

// ---------------------------------------------------------------------------
// Problem constants (from reference)
// ---------------------------------------------------------------------------
#define BB     8
#define NROI   400
#define DFEAT  512
#define H1DIM  450
#define EMB    360
#define GG     25
#define OG     12           // (25-3)/2+1
#define TMAX   (OG*OG*OG)   // 1728
#define NH     4
#define HD     90
#define HDP    96           // padded head dim for WMMA (multiple of 32)
#define FFD    2048
#define DEPTH  2
#define QTILES (TMAX/16)    // 108
#define ROWS   (BB*TMAX)    // 13824

typedef __attribute__((ext_vector_type(16))) _Float16 v16h;
typedef __attribute__((ext_vector_type(8)))  _Float16 v8h;
typedef __attribute__((ext_vector_type(2)))  _Float16 v2h;
typedef __attribute__((ext_vector_type(8)))  float    v8f;

__device__ __forceinline__ float gelu_exact(float x) {
    return 0.5f * x * (1.0f + erff(x * 0.70710678118654752440f));
}

// v_cvt_pk_rz_f16_f32 returns a __fp16 vector; bit-cast to our _Float16 pair
__device__ __forceinline__ v2h pk_f16(float x, float y) {
    auto t = __builtin_amdgcn_cvt_pkrtz(x, y);
    return __builtin_bit_cast(v2h, t);
}

__device__ __forceinline__ v16h cat16(v8h lo, v8h hi) {
    return __builtin_shufflevector(lo, hi, 0, 1, 2, 3, 4, 5, 6, 7,
                                           8, 9, 10, 11, 12, 13, 14, 15);
}

__device__ __forceinline__ v8f wmma_f16(const v16h& a, const v16h& b, const v8f& c) {
    return __builtin_amdgcn_wmma_f32_16x16x32_f16(false, a, false, b, (short)0, c, false, false);
}

__device__ __forceinline__ float half_reduce_max(float v) {
    #pragma unroll
    for (int o = 8; o >= 1; o >>= 1) v = fmaxf(v, __shfl_xor(v, o, 32));
    return v;
}
__device__ __forceinline__ float half_reduce_sum(float v) {
    #pragma unroll
    for (int o = 8; o >= 1; o >>= 1) v += __shfl_xor(v, o, 32);
    return v;
}

// ---------------------------------------------------------------------------
// fp32-in / f16-WMMA / fp32-out GEMM:  C = act(A(MxK) @ B(KxN) + bias)
// Block tile 64x128, K-step 32, 8 waves (2x4), each wave 32x32 (4 WMMAs).
// REQUIREMENT: M % 64 == 0, K even (true for all call sites here).
// Software-pipelined: global loads for tile k+1 issued during tile-k compute.
// Fragments load as aligned ds_load_b128 pairs:
//   A frag: As[row][8h..8h+7]  ++ As[row][16+8h..+7]         (pitch 40 halfs)
//   B frag: BsT[col][8h..8h+7] ++ BsT[col][16+8h..+7] (B tile stored transposed) (pitch 40 halfs)
// ---------------------------------------------------------------------------
#define GBM 64
#define GBN 128
#define GBK 32
#define APITCH 40
#define BPITCH 40

__global__ __launch_bounds__(256) void gemm_f16_kernel(
    const float* __restrict__ A, const float* __restrict__ Bw,
    const float* __restrict__ bias, float* __restrict__ Cout,
    int M, int N, int K, int act)
{
    __shared__ __align__(16) _Float16 As[GBM][APITCH];
    __shared__ __align__(16) _Float16 BsT[GBN][BPITCH];

    const int tid  = threadIdx.x;
    const int lane = tid & 31;
    const int wave = tid >> 5;
    const int half = lane >> 4;
    const int mn   = lane & 15;
    const int wm   = wave >> 2;   // 0..1
    const int wn   = wave & 3;    // 0..3
    const int m0   = blockIdx.y * GBM;
    const int n0   = blockIdx.x * GBN;

    // fixed per-thread staging coordinates (fixed trip counts -> no execz chains)
    int ar[4], ac[4];
    #pragma unroll
    for (int u = 0; u < 4; ++u) { int i2 = tid + 256 * u; ar[u] = i2 >> 4; ac[u] = (i2 & 15) * 2; }
    int bn[8], bk[8];
    #pragma unroll
    for (int u = 0; u < 8; ++u) { int i2 = tid + 256 * u; bn[u] = i2 & 127; bk[u] = i2 >> 7; }

    float2 areg[4];
    float  b0reg[8], b1reg[8];

    auto load_tile = [&](int k0) {
        const bool tail = (k0 + GBK) > K;
        if (!tail) {
            #pragma unroll
            for (int u = 0; u < 4; ++u)
                areg[u] = *(const float2*)(A + (size_t)(m0 + ar[u]) * K + k0 + ac[u]);
            #pragma unroll
            for (int u = 0; u < 8; ++u) {
                int gr  = k0 + 2 * bk[u];
                int gcN = min(n0 + bn[u], N - 1);   // clamp: garbage only in store-guarded cols
                b0reg[u] = Bw[(size_t)gr * N + gcN];
                b1reg[u] = Bw[(size_t)(gr + 1) * N + gcN];
            }
        } else {
            #pragma unroll
            for (int u = 0; u < 4; ++u) {
                float2 v = {0.0f, 0.0f};
                if (k0 + ac[u] < K)   // K even -> pair is all-or-nothing
                    v = *(const float2*)(A + (size_t)(m0 + ar[u]) * K + k0 + ac[u]);
                areg[u] = v;
            }
            #pragma unroll
            for (int u = 0; u < 8; ++u) {
                int gr  = k0 + 2 * bk[u];
                int gcN = min(n0 + bn[u], N - 1);
                float v0 = 0.0f, v1 = 0.0f;
                if (gr < K) {                      // K even -> pair all-or-nothing
                    v0 = Bw[(size_t)gr * N + gcN];
                    v1 = Bw[(size_t)(gr + 1) * N + gcN];
                }
                b0reg[u] = v0;
                b1reg[u] = v1;
            }
        }
    };
    auto store_tile = [&]() {
        #pragma unroll
        for (int u = 0; u < 4; ++u)
            *(v2h*)&As[ar[u]][ac[u]] = pk_f16(areg[u].x, areg[u].y);
        #pragma unroll
        for (int u = 0; u < 8; ++u)
            *(v2h*)&BsT[bn[u]][2 * bk[u]] = pk_f16(b0reg[u], b1reg[u]);
    };

    v8f c[2][2];
    #pragma unroll
    for (int i = 0; i < 2; ++i)
        #pragma unroll
        for (int j = 0; j < 2; ++j)
            #pragma unroll
            for (int r = 0; r < 8; ++r) c[i][j][r] = 0.0f;

    load_tile(0);
    for (int k0 = 0; k0 < K; k0 += GBK) {
        store_tile();
        __syncthreads();
        if (k0 + GBK < K) load_tile(k0 + GBK);   // prefetch next tile during compute

        v16h af[2], bf[2];
        #pragma unroll
        for (int s = 0; s < 2; ++s) {
            int row = wm * 32 + s * 16 + mn;
            af[s] = cat16(*(const v8h*)&As[row][8 * half],
                          *(const v8h*)&As[row][16 + 8 * half]);
        }
        #pragma unroll
        for (int s = 0; s < 2; ++s) {
            int col = wn * 32 + s * 16 + mn;
            bf[s] = cat16(*(const v8h*)&BsT[col][8 * half],
                          *(const v8h*)&BsT[col][16 + 8 * half]);
        }
        c[0][0] = wmma_f16(af[0], bf[0], c[0][0]);
        c[0][1] = wmma_f16(af[0], bf[1], c[0][1]);
        c[1][0] = wmma_f16(af[1], bf[0], c[1][0]);
        c[1][1] = wmma_f16(af[1], bf[1], c[1][1]);
        __syncthreads();
    }

    // ---- epilogue: bias + activation, N-guarded store (M multiple of 64) ----
    #pragma unroll
    for (int sm = 0; sm < 2; ++sm) {
        #pragma unroll
        for (int sn = 0; sn < 2; ++sn) {
            int rbase = m0 + wm * 32 + sm * 16 + 8 * half;
            int gc    = n0 + wn * 32 + sn * 16 + mn;
            if (gc >= N) continue;
            float bv = bias[gc];
            #pragma unroll
            for (int r = 0; r < 8; ++r) {
                float v = c[sm][sn][r] + bv;
                if (act == 1) v = gelu_exact(v);
                Cout[(size_t)(rbase + r) * N + gc] = v;
            }
        }
    }
}

// ---------------------------------------------------------------------------
// Node construction: gather F_pad rows via C and 3x3x3 stride-2 window sum
// ---------------------------------------------------------------------------
__global__ __launch_bounds__(128) void nodes_kernel(
    const float* __restrict__ Femb, const int* __restrict__ C,
    float* __restrict__ nodes)
{
    int q = blockIdx.x;          // b*1728 + p
    int b = q / TMAX, p = q % TMAX;
    int ox = p / (OG * OG), oy = (p / OG) % OG, oz = p % OG;

    __shared__ int idx[27];
    int tid = threadIdx.x;
    if (tid < 27) {
        int dx = tid / 9, dy = (tid / 3) % 3, dz = tid % 3;
        int gx = 2 * ox + dx, gy = 2 * oy + dy, gz = 2 * oz + dz;
        idx[tid] = C[(((size_t)b * GG + gx) * GG + gy) * GG + gz];
    }
    __syncthreads();

    int e = tid * 4;             // 90 active threads, float4 per thread
    if (e < EMB) {
        float4 acc = {0.0f, 0.0f, 0.0f, 0.0f};
        #pragma unroll
        for (int t = 0; t < 27; ++t) {
            int ci = idx[t];
            if (ci > 0) {
                float4 v = *(const float4*)(Femb + ((size_t)b * NROI + (ci - 1)) * EMB + e);
                acc.x += v.x; acc.y += v.y; acc.z += v.z; acc.w += v.w;
            }
        }
        *(float4*)(nodes + (size_t)q * EMB + e) = acc;
    }
}

// ---------------------------------------------------------------------------
// Validity flags: flag[p] = (sum over b,e of |nodes[b,p,e]|) > 0
// ---------------------------------------------------------------------------
__global__ __launch_bounds__(256) void flags_kernel(
    const float* __restrict__ nodes, int* __restrict__ flags)
{
    int p = blockIdx.x, tid = threadIdx.x;
    __shared__ float red[256];
    float s = 0.0f;
    for (int i = tid; i < BB * (EMB / 4); i += 256) {
        int b = i / (EMB / 4), e4 = (i % (EMB / 4)) * 4;
        float4 v = *(const float4*)(nodes + ((size_t)b * TMAX + p) * EMB + e4);
        s += fabsf(v.x) + fabsf(v.y) + fabsf(v.z) + fabsf(v.w);
    }
    red[tid] = s;
    __syncthreads();
    for (int o = 128; o > 0; o >>= 1) {
        if (tid < o) red[tid] += red[tid + o];
        __syncthreads();
    }
    if (tid == 0) flags[p] = (red[0] > 0.0f) ? 1 : 0;
}

// Serial compaction (1728 elems -> trivial)
__global__ void scan_kernel(const int* __restrict__ flags,
                            int* __restrict__ valid, int* __restrict__ tc)
{
    if (threadIdx.x == 0 && blockIdx.x == 0) {
        int cnt = 0;
        for (int p = 0; p < TMAX; ++p)
            if (flags[p]) valid[cnt++] = p;
        *tc = cnt;
    }
}

// tokens: x[b, j<Tc] = nodes[b, valid[j]]; zero fill beyond Tc
__global__ __launch_bounds__(128) void gather_tokens_kernel(
    const float* __restrict__ nodes, const int* __restrict__ valid,
    const int* __restrict__ tcp, float* __restrict__ x)
{
    int q = blockIdx.x;
    int b = q / TMAX, j = q % TMAX;
    int Tc = *tcp;
    size_t dst = ((size_t)b * TMAX + j) * EMB;
    int e = threadIdx.x * 4;     // 90 active threads
    if (e >= EMB) return;
    if (j < Tc) {
        size_t src = ((size_t)b * TMAX + valid[j]) * EMB;
        *(float4*)(x + dst + e) = *(const float4*)(nodes + src + e);
    } else {
        float4 z = {0.0f, 0.0f, 0.0f, 0.0f};
        *(float4*)(x + dst + e) = z;
    }
}

// ---------------------------------------------------------------------------
// Flash attention: one wave per (b, head, 16-query tile). Online softmax.
// qkv layout: [B*TMAX, 3*EMB]; q cols [0,360), k [360,720), v [720,1080).
// Staging: branchless float2 loads (clamped in-bounds rows), chunked
// two-phase (load-all-then-store) for memory-level parallelism. Fragment
// builds are aligned ds_load_b128 pairs via transposed/padded LDS layouts.
// ---------------------------------------------------------------------------
#define QPITCH 104   // 96 + 8, rows 16B-aligned

__global__ __launch_bounds__(32) void attn_kernel(
    const float* __restrict__ qkv, float* __restrict__ attn,
    const int* __restrict__ tcp)
{
    __shared__ __align__(16) _Float16 Qs[16][QPITCH];
    __shared__ __align__(16) _Float16 Ks[32][QPITCH];
    __shared__ __align__(16) _Float16 VsT[HDP][40];  // [dim][token]
    __shared__ __align__(16) _Float16 Ps[16][40];

    const int Tc   = *tcp;
    const int bid  = blockIdx.x;
    const int qt   = bid % QTILES;
    const int bh   = bid / QTILES;
    const int h    = bh & (NH - 1);
    const int b    = bh >> 2;
    const int lane = threadIdx.x;
    const int half = lane >> 4;
    const int mn   = lane & 15;
    const int q0   = qt * 16;
    const float scale = 0.10540925533894598f; // 1/sqrt(90)

    // ---- stage Q tile (16 x 90, pre-scaled, zero-padded to 96) ----
    #pragma unroll
    for (int ch = 0; ch < 3; ++ch) {
        float2 qb[8];
        int rr[8], dd[8];
        #pragma unroll
        for (int u = 0; u < 8; ++u) {
            int i = lane + 32 * (ch * 8 + u);
            rr[u] = i / 48;
            dd[u] = (i % 48) * 2;
            float2 v = {0.0f, 0.0f};
            if (dd[u] < HD)   // dd <= 88 -> pair in-bounds
                v = *(const float2*)(qkv + ((size_t)b * TMAX + q0 + rr[u]) * (3 * EMB) + h * HD + dd[u]);
            qb[u] = v;
        }
        #pragma unroll
        for (int u = 0; u < 8; ++u)
            *(v2h*)&Qs[rr[u]][dd[u]] = pk_f16(qb[u].x * scale, qb[u].y * scale);
    }
    __syncthreads();

    v16h qa[3];
    #pragma unroll
    for (int s = 0; s < 3; ++s)
        qa[s] = cat16(*(const v8h*)&Qs[mn][s * 32 + 8 * half],
                      *(const v8h*)&Qs[mn][s * 32 + 16 + 8 * half]);

    v8f o[6];
    #pragma unroll
    for (int g = 0; g < 6; ++g)
        #pragma unroll
        for (int r = 0; r < 8; ++r) o[g][r] = 0.0f;
    float mrow[8], lrow[8];
    #pragma unroll
    for (int r = 0; r < 8; ++r) { mrow[r] = -INFINITY; lrow[r] = 0.0f; }

    const int nk = (Tc + 31) / 32;
    for (int kc = 0; kc < nk; ++kc) {
        const int kb = kc * 32;
        __syncthreads();
        // ---- stage K (row-major) and V (transposed): chunked two-phase ----
        {
            int tokc = min(kb + lane, TMAX - 1);   // always in-bounds; masked later
            const float* kr = qkv + ((size_t)b * TMAX + tokc) * (3 * EMB) + EMB + h * HD;
            const float* vr = kr + EMB;
            #pragma unroll
            for (int ch = 0; ch < 5; ++ch) {        // 5 chunks x 9 pairs = 45 pairs = 90 dims
                float2 kb2[9], vb2[9];
                #pragma unroll
                for (int u = 0; u < 9; ++u) {
                    int d2 = (ch * 9 + u) * 2;
                    kb2[u] = *(const float2*)(kr + d2);
                    vb2[u] = *(const float2*)(vr + d2);
                }
                #pragma unroll
                for (int u = 0; u < 9; ++u) {
                    int d2 = (ch * 9 + u) * 2;
                    *(v2h*)&Ks[lane][d2] = pk_f16(kb2[u].x, kb2[u].y);
                    VsT[d2][lane]     = (_Float16)vb2[u].x;
                    VsT[d2 + 1][lane] = (_Float16)vb2[u].y;
                }
            }
            #pragma unroll
            for (int d2 = HD; d2 < HDP; d2 += 2) {
                v2h z = {(_Float16)0.0f, (_Float16)0.0f};
                *(v2h*)&Ks[lane][d2] = z;
            }
            #pragma unroll
            for (int d = HD; d < HDP; ++d) VsT[d][lane] = (_Float16)0.0f;
        }
        __syncthreads();

        // ---- S = Q @ K^T  (16 x 32 in two 16x16 frags) ----
        v8f S[2];
        #pragma unroll
        for (int s = 0; s < 2; ++s) {
            #pragma unroll
            for (int r = 0; r < 8; ++r) S[s][r] = 0.0f;
            #pragma unroll
            for (int step = 0; step < 3; ++step) {
                v16h bfr = cat16(*(const v8h*)&Ks[s * 16 + mn][step * 32 + 8 * half],
                                 *(const v8h*)&Ks[s * 16 + mn][step * 32 + 16 + 8 * half]);
                S[s] = wmma_f16(qa[step], bfr, S[s]);
            }
            if (kb + s * 16 + mn >= Tc) {   // mask keys beyond Tc
                #pragma unroll
                for (int r = 0; r < 8; ++r) S[s][r] = -1.0e30f;
            }
        }

        // ---- online softmax (rows r + 8*half live across the 16-lane half) ----
        #pragma unroll
        for (int r = 0; r < 8; ++r) {
            float mx = half_reduce_max(fmaxf(S[0][r], S[1][r]));
            float nm = fmaxf(mrow[r], mx);
            float al = __expf(mrow[r] - nm);
            mrow[r] = nm;
            float p0 = __expf(S[0][r] - nm);
            float p1 = __expf(S[1][r] - nm);
            S[0][r] = p0;
            S[1][r] = p1;
            float rs = half_reduce_sum(p0 + p1);
            lrow[r] = lrow[r] * al + rs;
            #pragma unroll
            for (int g = 0; g < 6; ++g) o[g][r] *= al;
        }

        // ---- re-layout P (C-layout -> A-layout) through LDS ----
        #pragma unroll
        for (int r = 0; r < 8; ++r) {
            Ps[r + 8 * half][mn]      = (_Float16)S[0][r];
            Ps[r + 8 * half][16 + mn] = (_Float16)S[1][r];
        }
        __syncthreads();
        v16h pa = cat16(*(const v8h*)&Ps[mn][8 * half],
                        *(const v8h*)&Ps[mn][16 + 8 * half]);

        // ---- O += P @ V  (16x32 @ 32x96) ----
        #pragma unroll
        for (int g = 0; g < 6; ++g) {
            v16h vb = cat16(*(const v8h*)&VsT[g * 16 + mn][8 * half],
                            *(const v8h*)&VsT[g * 16 + mn][16 + 8 * half]);
            o[g] = wmma_f16(pa, vb, o[g]);
        }
    }

    // ---- normalize and store (zero for invalid rows / padded dims) ----
    #pragma unroll
    for (int g = 0; g < 6; ++g) {
        int d = g * 16 + mn;
        if (d >= HD) continue;
        #pragma unroll
        for (int r = 0; r < 8; ++r) {
            int tok = q0 + r + 8 * half;
            float val = 0.0f;
            if (tok < Tc && lrow[r] > 0.0f) val = o[g][r] / lrow[r];
            attn[((size_t)b * TMAX + tok) * EMB + h * HD + d] = val;
        }
    }
}

// ---------------------------------------------------------------------------
// x = LayerNorm(x + y) * s + b  (in-place on x); zero rows beyond Tc
// ---------------------------------------------------------------------------
__global__ __launch_bounds__(128) void add_ln_kernel(
    float* __restrict__ x, const float* __restrict__ y,
    const float* __restrict__ sc, const float* __restrict__ bt,
    const int* __restrict__ tcp)
{
    int row = blockIdx.x;
    int j = row % TMAX;
    int tid = threadIdx.x;
    int Tc = *tcp;
    size_t base = (size_t)row * EMB;
    int e = tid * 4;             // 90 active threads, float4 lanes

    __shared__ float buf[EMB];
    __shared__ float red[128];

    if (j >= Tc) {
        if (e < EMB) {
            float4 z = {0.0f, 0.0f, 0.0f, 0.0f};
            *(float4*)(x + base + e) = z;
        }
        return;
    }
    float lsum = 0.0f;
    if (e < EMB) {
        float4 xv = *(const float4*)(x + base + e);
        float4 yv = *(const float4*)(y + base + e);
        float4 v  = {xv.x + yv.x, xv.y + yv.y, xv.z + yv.z, xv.w + yv.w};
        *(float4*)(buf + e) = v;
        lsum = v.x + v.y + v.z + v.w;
    }
    red[tid] = lsum;
    __syncthreads();
    for (int o = 64; o > 0; o >>= 1) {
        if (tid < o) red[tid] += red[tid + o];
        __syncthreads();
    }
    float mean = red[0] * (1.0f / EMB);
    __syncthreads();
    float lvar = 0.0f;
    if (e < EMB) {
        #pragma unroll
        for (int t = 0; t < 4; ++t) {
            float d = buf[e + t] - mean;
            lvar += d * d;
        }
    }
    red[tid] = lvar;
    __syncthreads();
    for (int o = 64; o > 0; o >>= 1) {
        if (tid < o) red[tid] += red[tid + o];
        __syncthreads();
    }
    float rstd = rsqrtf(red[0] * (1.0f / EMB) + 1e-5f);
    __syncthreads();
    if (e < EMB) {
        float4 v = *(const float4*)(buf + e);
        float4 s4 = *(const float4*)(sc + e);
        float4 b4 = *(const float4*)(bt + e);
        float4 outv = {(v.x - mean) * rstd * s4.x + b4.x,
                       (v.y - mean) * rstd * s4.y + b4.y,
                       (v.z - mean) * rstd * s4.z + b4.z,
                       (v.w - mean) * rstd * s4.w + b4.w};
        *(float4*)(x + base + e) = outv;
    }
}

// ---------------------------------------------------------------------------
// Final head: mean over valid tokens -> 360->256 GELU -> 128 ReLU -> 2
// ---------------------------------------------------------------------------
__global__ __launch_bounds__(256) void head_kernel(
    const float* __restrict__ x, const int* __restrict__ tcp,
    const float* __restrict__ cw1, const float* __restrict__ cb1,
    const float* __restrict__ cw2, const float* __restrict__ cb2,
    const float* __restrict__ cw3, const float* __restrict__ cb3,
    float* __restrict__ out)
{
    int b = blockIdx.x, tid = threadIdx.x;
    int Tc = *tcp;
    __shared__ float h[EMB];
    __shared__ float h1[256];
    __shared__ float h2[128];

    for (int e = tid; e < EMB; e += 256) {
        float s = 0.0f;
        for (int j = 0; j < Tc; ++j) s += x[((size_t)b * TMAX + j) * EMB + e];
        h[e] = (Tc > 0) ? s / (float)Tc : 0.0f;
    }
    __syncthreads();
    {
        float s = cb1[tid];
        for (int e = 0; e < EMB; ++e) s += h[e] * cw1[(size_t)e * 256 + tid];
        h1[tid] = gelu_exact(s);
    }
    __syncthreads();
    if (tid < 128) {
        float s = cb2[tid];
        for (int e = 0; e < 256; ++e) s += h1[e] * cw2[(size_t)e * 128 + tid];
        h2[tid] = fmaxf(s, 0.0f);
    }
    __syncthreads();
    if (tid < 2) {
        float s = cb3[tid];
        for (int e = 0; e < 128; ++e) s += h2[e] * cw3[(size_t)e * 2 + tid];
        out[b * 2 + tid] = s;
    }
}

// ---------------------------------------------------------------------------
// Host orchestration
// ---------------------------------------------------------------------------
extern "C" void kernel_launch(void* const* d_in, const int* in_sizes, int n_in,
                              void* d_out, int out_size, void* d_ws, size_t ws_size,
                              hipStream_t stream) {
    const float* F_roi  = (const float*)d_in[0];
    const int*   Cidx   = (const int*)  d_in[1];
    const float* ffn_w1 = (const float*)d_in[2];
    const float* ffn_b1 = (const float*)d_in[3];
    const float* ffn_w2 = (const float*)d_in[4];
    const float* ffn_b2 = (const float*)d_in[5];
    const float* wqkv   = (const float*)d_in[6];
    const float* bqkv   = (const float*)d_in[7];
    const float* wo     = (const float*)d_in[8];
    const float* bo     = (const float*)d_in[9];
    const float* ln1_s  = (const float*)d_in[10];
    const float* ln1_b  = (const float*)d_in[11];
    const float* wf1    = (const float*)d_in[12];
    const float* bf1    = (const float*)d_in[13];
    const float* wf2    = (const float*)d_in[14];
    const float* bf2    = (const float*)d_in[15];
    const float* ln2_s  = (const float*)d_in[16];
    const float* ln2_b  = (const float*)d_in[17];
    const float* cw1    = (const float*)d_in[18];
    const float* cb1    = (const float*)d_in[19];
    const float* cw2    = (const float*)d_in[20];
    const float* cb2    = (const float*)d_in[21];
    const float* cw3    = (const float*)d_in[22];
    const float* cb3    = (const float*)d_in[23];

    float* ws = (float*)d_ws;
    const size_t S_X   = (size_t)ROWS * EMB;      //  4,976,640
    const size_t S_QKV = (size_t)ROWS * 3 * EMB;  // 14,929,920
    const size_t S_FFN = (size_t)ROWS * FFD;      // 28,311,552

    float* x_buf  = ws;
    float* qkv_b  = x_buf + S_X;
    // node-stage buffers alias the (later-written) qkv region
    float* nodes  = qkv_b;                                  // ROWS*EMB
    float* emb    = nodes + S_X;                            // 3200*360
    float* h1b    = emb + (size_t)BB * NROI * EMB;          // 3200*450
    float* attn_b = qkv_b + S_QKV;
    float* proj_b = attn_b + S_X;
    float* ffn_b  = proj_b + S_X;
    int*   ibuf   = (int*)(ffn_b + S_FFN);
    int*   flags  = ibuf;
    int*   valid  = ibuf + TMAX;
    int*   tc     = ibuf + 2 * TMAX;

    auto gemm = [&](const float* A, const float* Bw, const float* bias,
                    float* Co, int M, int N, int K, int act) {
        dim3 grid((N + GBN - 1) / GBN, (M + GBM - 1) / GBM);
        gemm_f16_kernel<<<grid, dim3(256), 0, stream>>>(A, Bw, bias, Co, M, N, K, act);
    };

    // ---- node embedding FFN ----
    gemm(F_roi, ffn_w1, ffn_b1, h1b, BB * NROI, H1DIM, DFEAT, 1);
    gemm(h1b,   ffn_w2, ffn_b2, emb, BB * NROI, EMB,   H1DIM, 0);

    // ---- gather + 3x3x3/2 window sum, mask, compact ----
    nodes_kernel<<<BB * TMAX, 128, 0, stream>>>(emb, Cidx, nodes);
    flags_kernel<<<TMAX, 256, 0, stream>>>(nodes, flags);
    scan_kernel<<<1, 32, 0, stream>>>(flags, valid, tc);
    gather_tokens_kernel<<<BB * TMAX, 128, 0, stream>>>(nodes, valid, tc, x_buf);

    // ---- transformer encoder ----
    for (int l = 0; l < DEPTH; ++l) {
        gemm(x_buf, wqkv + (size_t)l * EMB * 3 * EMB, bqkv + (size_t)l * 3 * EMB,
             qkv_b, ROWS, 3 * EMB, EMB, 0);
        attn_kernel<<<BB * NH * QTILES, 32, 0, stream>>>(qkv_b, attn_b, tc);
        gemm(attn_b, wo + (size_t)l * EMB * EMB, bo + (size_t)l * EMB,
             proj_b, ROWS, EMB, EMB, 0);
        add_ln_kernel<<<ROWS, 128, 0, stream>>>(x_buf, proj_b,
             ln1_s + (size_t)l * EMB, ln1_b + (size_t)l * EMB, tc);
        gemm(x_buf, wf1 + (size_t)l * EMB * FFD, bf1 + (size_t)l * FFD,
             ffn_b, ROWS, FFD, EMB, 1);
        gemm(ffn_b, wf2 + (size_t)l * FFD * EMB, bf2 + (size_t)l * EMB,
             proj_b, ROWS, EMB, FFD, 0);
        add_ln_kernel<<<ROWS, 128, 0, stream>>>(x_buf, proj_b,
             ln2_s + (size_t)l * EMB, ln2_b + (size_t)l * EMB, tc);
    }

    // ---- pooled classifier head ----
    head_kernel<<<BB, 256, 0, stream>>>(x_buf, tc, cw1, cb1, cw2, cb2, cw3, cb3,
                                        (float*)d_out);
}
